// ModularGraphAttentionTransformer_89824946028990
// MI455X (gfx1250) — matrix operand order repaired and checked by reading, not compiled, not run on hardware
//
#include <hip/hip_runtime.h>
#include <math.h>

// ---------------------------------------------------------------------------
// ModularGraphAttentionTransformer forward for gfx1250 (MI455X, wave32, WMMA)
//
// GEMMs (N x 128 @ 128 x 128) run on V_WMMA_F32_16X16X32_F16 with f32
// accumulate: activations converted to f16, weights transposed+converted so
// A and B wave fragments are contiguous b128 loads per the CDNA5 VGPR layout.
// Edge softmax-aggregation is done with the separable-normalizer trick
// (accumulate exp-weighted messages + denom, divide once per node) to avoid
// materializing per-edge alpha. Whole working set (~100MB) fits the 192MB L2.
// ---------------------------------------------------------------------------

#define HID   128
#define HEADS 8
#define CH    16
#define NGR   256   // num_graphs (reference constant G)

typedef __attribute__((ext_vector_type(16))) _Float16 v16h;
typedef __attribute__((ext_vector_type(8)))  float    v8f;

__device__ __forceinline__ float gelu_f(float x) {
  // exact GELU: 0.5*x*(1+erf(x/sqrt(2)))
  return 0.5f * x * (1.0f + erff(x * 0.70710678118654752440f));
}

__device__ __forceinline__ void atomic_max_f32(float* addr, float val) {
  unsigned int* ua = reinterpret_cast<unsigned int*>(addr);
  unsigned int old = __float_as_uint(*addr);
  while (__uint_as_float(old) < val) {
    unsigned int assumed = old;
    old = atomicCAS(ua, assumed, __float_as_uint(val));
    if (old == assumed) break;
  }
}

// ---------------- elementwise helpers ----------------

__global__ __launch_bounds__(256) void k_f32_to_f16(const float* __restrict__ in,
                                                    _Float16* __restrict__ out, int n) {
  int t = blockIdx.x * 256 + threadIdx.x;
  if (t < n) out[t] = (_Float16)in[t];
}

// W: [K, 128] row-major (f32)  ->  WT: [128, K] row-major (f16)
__global__ __launch_bounds__(256) void k_wtrans_f16(const float* __restrict__ W,
                                                    _Float16* __restrict__ WT,
                                                    int K, int n) {
  int t = blockIdx.x * 256 + threadIdx.x;
  if (t >= n) return;
  int col = t / K, k = t % K;
  WT[(size_t)col * K + k] = (_Float16)W[(size_t)k * HID + col];
}

__global__ __launch_bounds__(256) void k_fill(float* __restrict__ p, float v, int n) {
  int t = blockIdx.x * 256 + threadIdx.x;
  if (t < n) p[t] = v;
}

// ---------------- WMMA GEMM: out[M,128] = A[M,K](f16) * B[K,128] ----------------
// BT is B transposed: [128, K] f16.  One block = 8 waves = 16-row x 128-col strip.
// Wave fragment layouts (wave32, 16-bit data, CDNA5 ISA 7.12.2):
//   A 16x32: lane l -> row m=l&15; halves e=0..7 at K=kk+hi*8+e, e=8..15 at K=kk+16+hi*8+(e-8)
//   B 32x16: lane l -> col n=l&15; same K mapping as A (e=0..7 at kk+hi*8+e, e=8..15 at kk+16+hi*8+(e-8))
//   C 16x16: lane l, vgpr r -> M = hi*8 + r, N = l&15
// mode 0: plain store; mode 1: gelu(v + bias[col])
template <int K>
__global__ __launch_bounds__(256) void k_gemm_wmma(const _Float16* __restrict__ A,
                                                   const _Float16* __restrict__ BT,
                                                   const float* __restrict__ bias,
                                                   float* __restrict__ out, int mode) {
  const int lane = threadIdx.x & 31;
  const int wv   = threadIdx.x >> 5;   // column tile 0..7
  const int m    = lane & 15;
  const int hi   = lane >> 4;
  const int rowBase = blockIdx.x << 4;
  const int colBase = wv << 4;

  v8f acc = {0.f, 0.f, 0.f, 0.f, 0.f, 0.f, 0.f, 0.f};
  const _Float16* arow = A  + (size_t)(rowBase + m) * K + hi * 8;
  const _Float16* brow = BT + (size_t)(colBase + m) * K + hi * 8;

#pragma unroll
  for (int kk = 0; kk < K; kk += 32) {
    v16h a, b;
    ((uint4*)&a)[0] = *(const uint4*)(arow + kk);        // K = kk+hi*8   .. +7
    ((uint4*)&a)[1] = *(const uint4*)(arow + kk + 16);   // K = kk+16+hi*8.. +7
    ((uint4*)&b)[0] = *(const uint4*)(brow + kk);        // K = kk+hi*8   .. +7
    ((uint4*)&b)[1] = *(const uint4*)(brow + kk + 16);   // K = kk+16+hi*8.. +7
    acc = __builtin_amdgcn_wmma_f32_16x16x32_f16(false, a, false, b,
                                                 (short)0, acc, false, false);
  }

#pragma unroll
  for (int r = 0; r < 8; ++r) {
    int row = rowBase + hi * 8 + r;
    int col = colBase + m;
    float v = acc[r];
    if (mode == 1) v = gelu_f(v + bias[col]);
    out[(size_t)row * HID + col] = v;
  }
}

// ---------------- BatchNorm (training-mode biased stats) ----------------

__global__ __launch_bounds__(256) void k_bn_stats(const float* __restrict__ X,
                                                  float* __restrict__ mu,
                                                  float* __restrict__ rsig, int n_rows) {
  __shared__ float sh[256], sh2[256];
  int c = blockIdx.x;  // one block per channel
  float s = 0.f, s2 = 0.f;
  for (int n = threadIdx.x; n < n_rows; n += 256) {
    float v = X[(size_t)n * HID + c];
    s += v; s2 += v * v;
  }
  sh[threadIdx.x] = s; sh2[threadIdx.x] = s2;
  __syncthreads();
  for (int off = 128; off > 0; off >>= 1) {
    if (threadIdx.x < off) {
      sh[threadIdx.x]  += sh[threadIdx.x + off];
      sh2[threadIdx.x] += sh2[threadIdx.x + off];
    }
    __syncthreads();
  }
  if (threadIdx.x == 0) {
    float mean = sh[0] / n_rows;
    float var  = sh2[0] / n_rows - mean * mean;
    mu[c]   = mean;
    rsig[c] = rsqrtf(var + 1e-5f);
  }
}

__global__ __launch_bounds__(256) void k_bn_affine(float* __restrict__ X,
                                                   const float* __restrict__ mu,
                                                   const float* __restrict__ rsig,
                                                   const float* __restrict__ g,
                                                   const float* __restrict__ b, int n) {
  int t = blockIdx.x * 256 + threadIdx.x;
  if (t >= n) return;
  int c = t & (HID - 1);
  X[t] = g[c] * (X[t] - mu[c]) * rsig[c] + b[c];
}

// h = gelu(bn(acc)) + h   (residual update)
__global__ __launch_bounds__(256) void k_layer_update(float* __restrict__ h,
                                                      const float* __restrict__ acc,
                                                      const float* __restrict__ mu,
                                                      const float* __restrict__ rsig,
                                                      const float* __restrict__ g,
                                                      const float* __restrict__ b, int n) {
  int t = blockIdx.x * 256 + threadIdx.x;
  if (t >= n) return;
  int c = t & (HID - 1);
  float v = g[c] * (acc[t] - mu[c]) * rsig[c] + b[c];
  h[t] = gelu_f(v) + h[t];
}

// ---------------- GAT attention ----------------

// per-(node, head) attention scores a_src/a_dst = <xt[n,h,:], att_*[h,:]>
__global__ __launch_bounds__(256) void k_att_scores(const float* __restrict__ xt,
                                                    const float* __restrict__ att_src,
                                                    const float* __restrict__ att_dst,
                                                    float* __restrict__ a_src,
                                                    float* __restrict__ a_dst, int n) {
  int t = blockIdx.x * 256 + threadIdx.x;
  if (t >= n) return;
  int node = t >> 3, hh = t & 7;
  const float* xp = xt + (size_t)node * HID + hh * CH;
  const float* as = att_src + hh * CH;
  const float* ad = att_dst + hh * CH;
  float s1 = 0.f, s2 = 0.f;
#pragma unroll
  for (int c = 0; c < CH; ++c) { float v = xp[c]; s1 += v * as[c]; s2 += v * ad[c]; }
  a_src[t] = s1; a_dst[t] = s2;
}

__device__ __forceinline__ void edge_decode(const int* __restrict__ ei, int E, int idx,
                                            int& s, int& d) {
  if (idx < E) { s = ei[idx]; d = ei[E + idx]; }
  else         { s = d = idx - E; }           // implicit self loops appended
}

// pass 1: per-dest per-head segment max of leaky_relu(a_src[src]+a_dst[dst])
__global__ __launch_bounds__(256) void k_edge_max(const int* __restrict__ ei, int E, int Nn,
                                                  const float* __restrict__ a_src,
                                                  const float* __restrict__ a_dst,
                                                  float* __restrict__ segmax) {
  int t = blockIdx.x * 256 + threadIdx.x;
  int total = (E + Nn) * HEADS;
  if (t >= total) return;
  int idx = t >> 3, hh = t & 7, s, d;
  edge_decode(ei, E, idx, s, d);
  float e = a_src[s * HEADS + hh] + a_dst[d * HEADS + hh];
  e = (e > 0.f) ? e : 0.2f * e;
  atomic_max_f32(&segmax[d * HEADS + hh], e);
}

// pass 2: accumulate exp-weighted messages and denominators (separable softmax)
__global__ __launch_bounds__(256) void k_edge_acc(const int* __restrict__ ei, int E, int Nn,
                                                  const float* __restrict__ a_src,
                                                  const float* __restrict__ a_dst,
                                                  const float* __restrict__ segmax,
                                                  const float* __restrict__ xt,
                                                  float* __restrict__ denom,
                                                  float* __restrict__ accum) {
  int t = blockIdx.x * 256 + threadIdx.x;
  int total = (E + Nn) * HEADS;
  if (t >= total) return;
  int idx = t >> 3, hh = t & 7, s, d;
  edge_decode(ei, E, idx, s, d);
  float e = a_src[s * HEADS + hh] + a_dst[d * HEADS + hh];
  e = (e > 0.f) ? e : 0.2f * e;
  float ex = expf(e - segmax[d * HEADS + hh]);
  atomicAdd(&denom[d * HEADS + hh], ex);
  const float* xs = xt + (size_t)s * HID + hh * CH;
  float* ap = accum + (size_t)d * HID + hh * CH;
#pragma unroll
  for (int c = 0; c < CH; ++c) atomicAdd(&ap[c], ex * xs[c]);
}

// pass 3: normalize by denom and add layer bias
__global__ __launch_bounds__(256) void k_gat_finish(float* __restrict__ accum,
                                                    const float* __restrict__ denom,
                                                    const float* __restrict__ bias, int n) {
  int t = blockIdx.x * 256 + threadIdx.x;
  if (t >= n) return;
  int node = t / HID, c = t & (HID - 1), hh = c >> 4;
  accum[t] = accum[t] / (denom[node * HEADS + hh] + 1e-16f) + bias[c];
}

// ---------------- attentional graph pooling ----------------

__global__ __launch_bounds__(256) void k_gate(const float* __restrict__ g1,
                                              const float* __restrict__ Wg2,
                                              const float* __restrict__ bg2,
                                              float* __restrict__ gate, int Nn) {
  int node = blockIdx.x * 256 + threadIdx.x;
  if (node >= Nn) return;
  float s = bg2[0];
  const float* r = g1 + (size_t)node * HID;
#pragma unroll 8
  for (int c = 0; c < HID; ++c) s += r[c] * Wg2[c];
  gate[node] = s;
}

__global__ __launch_bounds__(256) void k_graph_max(const float* __restrict__ gate,
                                                   const int* __restrict__ batch,
                                                   float* __restrict__ gm, int Nn) {
  int t = blockIdx.x * 256 + threadIdx.x;
  if (t < Nn) atomic_max_f32(&gm[batch[t]], gate[t]);
}

__global__ __launch_bounds__(256) void k_graph_acc(const float* __restrict__ gate,
                                                   const int* __restrict__ batch,
                                                   const float* __restrict__ gm,
                                                   const float* __restrict__ h,
                                                   float* __restrict__ gs,
                                                   float* __restrict__ pooled, int n) {
  int t = blockIdx.x * 256 + threadIdx.x;
  if (t >= n) return;
  int node = t / HID, c = t & (HID - 1);
  int bg = batch[node];
  float ge = expf(gate[node] - gm[bg]);
  atomicAdd(&pooled[(size_t)bg * HID + c], ge * h[t]);
  if (c == 0) atomicAdd(&gs[bg], ge);
}

__global__ __launch_bounds__(256) void k_graph_norm(float* __restrict__ pooled,
                                                    const float* __restrict__ gs, int n) {
  int t = blockIdx.x * 256 + threadIdx.x;
  if (t >= n) return;
  int g = t / HID;
  pooled[t] = pooled[t] / (gs[g] + 1e-16f);
}

// ---------------- output MLP (tiny: 256x128 @ 128x256 -> 256x1) ----------------

__global__ __launch_bounds__(256) void k_mlp1(const float* __restrict__ pooled,
                                              const float* __restrict__ Wo1,
                                              const float* __restrict__ bo1,
                                              float* __restrict__ q, int n) {
  int t = blockIdx.x * 256 + threadIdx.x;
  if (t >= n) return;
  int g = t >> 8, j = t & 255;
  float s = bo1[j];
  const float* pr = pooled + (size_t)g * HID;
#pragma unroll 8
  for (int k = 0; k < HID; ++k) s += pr[k] * Wo1[(size_t)k * 256 + j];
  q[t] = gelu_f(s);
}

__global__ __launch_bounds__(256) void k_mlp2(const float* __restrict__ q,
                                              const float* __restrict__ Wo2,
                                              const float* __restrict__ bo2,
                                              float* __restrict__ out) {
  int g = threadIdx.x;  // 256 graphs, one block
  float s = bo2[0];
  const float* qr = q + (size_t)g * 256;
#pragma unroll 8
  for (int k = 0; k < 256; ++k) s += qr[k] * Wo2[k];
  out[g] = s;
}

// ---------------------------------------------------------------------------

extern "C" void kernel_launch(void* const* d_in, const int* in_sizes, int n_in,
                              void* d_out, int out_size, void* d_ws, size_t ws_size,
                              hipStream_t stream) {
  (void)n_in; (void)out_size; (void)ws_size;

  const float* x      = (const float*)d_in[0];
  const int*   ei     = (const int*)d_in[1];
  const int*   batch  = (const int*)d_in[2];
  // d_in[3] = num_graphs (device scalar); G is the reference constant 256.
  const float* W_emb  = (const float*)d_in[4];
  const float* b_emb  = (const float*)d_in[5];
  const float* g_emb  = (const float*)d_in[6];
  const float* be_emb = (const float*)d_in[7];
  // layers at 8 + 6*l: W, att_src, att_dst, bias, g, b
  const float* Wg1 = (const float*)d_in[32];
  const float* bg1 = (const float*)d_in[33];
  const float* Wg2 = (const float*)d_in[34];
  const float* bg2 = (const float*)d_in[35];
  const float* Wo1 = (const float*)d_in[36];
  const float* bo1 = (const float*)d_in[37];
  const float* Wo2 = (const float*)d_in[38];
  const float* bo2 = (const float*)d_in[39];

  const int Nn = in_sizes[0] / 64;   // 50000 (divisible by 16)
  const int E  = in_sizes[1] / 2;    // 1,600,000
  const int NH = Nn * HID;

  // --- carve workspace (everything 256B aligned; ~97 MB total) ---
  char* base = (char*)d_ws;
  size_t off = 0;
  auto carve = [&](size_t bytes) -> char* {
    char* p = base + off;
    off += (bytes + 255) & ~(size_t)255;
    return p;
  };
  float*    h      = (float*)carve((size_t)NH * 4);
  float*    accb   = (float*)carve((size_t)NH * 4);
  float*    xt     = (float*)carve((size_t)NH * 4);
  _Float16* h16    = (_Float16*)carve((size_t)NH * 2);   // also holds x16 (N*64)
  _Float16* wt16   = (_Float16*)carve((size_t)HID * HID * 2);
  float*    a_src  = (float*)carve((size_t)Nn * HEADS * 4);
  float*    a_dst  = (float*)carve((size_t)Nn * HEADS * 4);
  float*    segmax = (float*)carve((size_t)Nn * HEADS * 4);
  float*    denom  = (float*)carve((size_t)Nn * HEADS * 4);
  float*    gate   = (float*)carve((size_t)Nn * 4);
  float*    gm     = (float*)carve(NGR * 4);
  float*    gs     = (float*)carve(NGR * 4);
  float*    pooled = (float*)carve((size_t)NGR * HID * 4);
  float*    q      = (float*)carve((size_t)NGR * 256 * 4);
  float*    mu     = (float*)carve(HID * 4);
  float*    rsig   = (float*)carve(HID * 4);

  dim3 blk(256);
  auto g1d = [](int n) { return dim3((unsigned)((n + 255) / 256)); };
  const int rowTiles = Nn / 16;            // 3125 WMMA row strips
  const int totE     = (E + Nn) * HEADS;   // edges + self loops, per head

  // --- embedding: h = bn(gelu(x @ W_emb + b_emb)) ---
  k_f32_to_f16<<<g1d(Nn * 64), blk, 0, stream>>>(x, h16, Nn * 64);
  k_wtrans_f16<<<g1d(HID * 64), blk, 0, stream>>>(W_emb, wt16, 64, HID * 64);
  k_gemm_wmma<64><<<rowTiles, blk, 0, stream>>>(h16, wt16, b_emb, h, 1);
  k_bn_stats<<<HID, blk, 0, stream>>>(h, mu, rsig, Nn);
  k_bn_affine<<<g1d(NH), blk, 0, stream>>>(h, mu, rsig, g_emb, be_emb, NH);

  // --- 4 GAT layers ---
  for (int l = 0; l < 4; ++l) {
    const float* Wl     = (const float*)d_in[8 + 6 * l + 0];
    const float* att_s  = (const float*)d_in[8 + 6 * l + 1];
    const float* att_d  = (const float*)d_in[8 + 6 * l + 2];
    const float* bias_l = (const float*)d_in[8 + 6 * l + 3];
    const float* g_l    = (const float*)d_in[8 + 6 * l + 4];
    const float* b_l    = (const float*)d_in[8 + 6 * l + 5];

    k_f32_to_f16<<<g1d(NH), blk, 0, stream>>>(h, h16, NH);
    k_wtrans_f16<<<g1d(HID * HID), blk, 0, stream>>>(Wl, wt16, HID, HID * HID);
    k_gemm_wmma<128><<<rowTiles, blk, 0, stream>>>(h16, wt16, nullptr, xt, 0);

    k_att_scores<<<g1d(Nn * HEADS), blk, 0, stream>>>(xt, att_s, att_d, a_src, a_dst,
                                                      Nn * HEADS);
    k_fill<<<g1d(Nn * HEADS), blk, 0, stream>>>(segmax, -3.0e38f, Nn * HEADS);
    k_fill<<<g1d(Nn * HEADS), blk, 0, stream>>>(denom, 0.f, Nn * HEADS);
    k_fill<<<g1d(NH), blk, 0, stream>>>(accb, 0.f, NH);
    k_edge_max<<<g1d(totE), blk, 0, stream>>>(ei, E, Nn, a_src, a_dst, segmax);
    k_edge_acc<<<g1d(totE), blk, 0, stream>>>(ei, E, Nn, a_src, a_dst, segmax, xt,
                                              denom, accb);
    k_gat_finish<<<g1d(NH), blk, 0, stream>>>(accb, denom, bias_l, NH);

    k_bn_stats<<<HID, blk, 0, stream>>>(accb, mu, rsig, Nn);
    k_layer_update<<<g1d(NH), blk, 0, stream>>>(h, accb, mu, rsig, g_l, b_l, NH);
  }

  // --- attentional aggregation ---
  k_f32_to_f16<<<g1d(NH), blk, 0, stream>>>(h, h16, NH);
  k_wtrans_f16<<<g1d(HID * HID), blk, 0, stream>>>(Wg1, wt16, HID, HID * HID);
  k_gemm_wmma<128><<<rowTiles, blk, 0, stream>>>(h16, wt16, bg1, xt, 1);  // gelu(h@Wg1+bg1)
  k_gate<<<g1d(Nn), blk, 0, stream>>>(xt, Wg2, bg2, gate, Nn);

  k_fill<<<g1d(NGR), blk, 0, stream>>>(gm, -3.0e38f, NGR);
  k_fill<<<g1d(NGR), blk, 0, stream>>>(gs, 0.f, NGR);
  k_fill<<<g1d(NGR * HID), blk, 0, stream>>>(pooled, 0.f, NGR * HID);
  k_graph_max<<<g1d(Nn), blk, 0, stream>>>(gate, batch, gm, Nn);
  k_graph_acc<<<g1d(NH), blk, 0, stream>>>(gate, batch, gm, h, gs, pooled, NH);
  k_graph_norm<<<g1d(NGR * HID), blk, 0, stream>>>(pooled, gs, NGR * HID);

  // --- output MLP ---
  k_mlp1<<<g1d(NGR * 256), blk, 0, stream>>>(pooled, Wo1, bo1, q, NGR * 256);
  k_mlp2<<<1, blk, 0, stream>>>(q, Wo2, bo2, (float*)d_out);
}
